// MultiHeadedGraphAttention_44882408243580
// MI455X (gfx1250) — compile-verified, matrix operand order checked by reading, not run on hardware
//
#include <hip/hip_runtime.h>
#include <hip/hip_bf16.h>

// ---------------------------------------------------------------------------
// MultiHeadedGraphAttention for MI455X (gfx1250, wave32, WMMA, async-LDS).
// B=8, M=1024, D=512, H=8, HD=64.
// Key identity: scores = Q(K+P)^T, ctx = attn(V+P) with P = pe.sum(0).
// ---------------------------------------------------------------------------

typedef __attribute__((ext_vector_type(16))) _Float16 v16h;
typedef __attribute__((ext_vector_type(8)))  _Float16 v8h;
typedef __attribute__((ext_vector_type(4)))  _Float16 v4h;
typedef __attribute__((ext_vector_type(8)))  float    v8f;
typedef __attribute__((ext_vector_type(4)))  float    v4f;

union H16 { v16h v; v8h h[2]; };

#define WMMA_F16(a, b, c) \
  __builtin_amdgcn_wmma_f32_16x16x32_f16(false, (a), false, (b), (short)0, (c), false, false)

#define BB   8
#define MM   1024
#define DD   512
#define HH   8
#define HDD  64

// Async global->LDS 16-byte copy (gfx1250 GLOBAL_LOAD_ASYNC_TO_LDS_B128, GV
// mode). The generic pointer to a __shared__ object carries the LDS byte
// offset in its low 32 bits.
__device__ __forceinline__ void async_lds_b128(void* lds_dst, const void* gsrc) {
  unsigned lds = (unsigned)(unsigned long long)lds_dst;
  unsigned long long ga = (unsigned long long)gsrc;
  asm volatile("global_load_async_to_lds_b128 %0, %1, off"
               :: "v"(lds), "v"(ga) : "memory");
}

// ---------------------------------------------------------------------------
// Kernel 1: P[j][d] = sum_i pe[i][j][d].  256 MB streamed once: non-temporal
// b128 loads (don't pollute the 192 MB L2) + software prefetch.
// ---------------------------------------------------------------------------
__global__ __launch_bounds__(256) void pe_reduce_kernel(
    const float* __restrict__ pe, float* __restrict__ P)
{
  const int t4 = (blockIdx.x * 256 + threadIdx.x) * 4;   // flat (j,d)*4
  const v4f* p = (const v4f*)(pe + t4);
  const size_t stride = (size_t)MM * HDD / 4;            // rows of pe, in v4f
  v4f acc = {};
  for (int i = 0; i < MM; ++i) {
    const int ip = (i + 8 < MM) ? i + 8 : MM - 1;
    __builtin_prefetch(&p[(size_t)ip * stride], 0, 1);
    v4f x = __builtin_nontemporal_load(&p[(size_t)i * stride]);
    acc += x;
  }
  *(v4f*)(P + t4) = acc;
}

// ---------------------------------------------------------------------------
// Kernel 2: projection GEMM  Y = X[8192,512] @ W[512,512] + bias.
// MODE: 0 = Q (scale 1/8), 1 = K (+P), 2 = V (+P).  f16 out in [B,H,M,HD].
// Block: 256 threads = 8 waves; block tile 256x64; wave tile 32x64 (8 WMMA/k).
// ---------------------------------------------------------------------------
template <int MODE>
__global__ __launch_bounds__(256) void proj_kernel(
    const float* __restrict__ X, const float* __restrict__ W,
    const float* __restrict__ bias, const float* __restrict__ P,
    _Float16* __restrict__ Y)
{
  __shared__ _Float16 Ash[256][32];   // A tile, row-major (rows x k)
  __shared__ _Float16 Bsh[64][32];    // W tile, transposed (n x k)

  const int tid  = threadIdx.x;
  const int w    = tid >> 5;
  const int lane = tid & 31;
  const int r    = lane & 15;
  const int hi   = lane >> 4;
  const int r0   = blockIdx.x * 256;
  const int n0   = blockIdx.y * 64;

  v8f acc[2][4] = {{v8f{}, v8f{}, v8f{}, v8f{}}, {v8f{}, v8f{}, v8f{}, v8f{}}};

  for (int k0 = 0; k0 < DD; k0 += 32) {
    // --- stage A: one 32-float row slice per thread, f32 -> f16 ---
    {
      const float* xr = X + (size_t)(r0 + tid) * DD + k0;
      #pragma unroll
      for (int j = 0; j < 8; ++j) {
        v4f f = *(const v4f*)(xr + 4 * j);
        v4h h4 = { (_Float16)f[0], (_Float16)f[1], (_Float16)f[2], (_Float16)f[3] };
        *(v4h*)(&Ash[tid][4 * j]) = h4;
      }
    }
    // --- stage B (transposed): Bsh[n][kk] = W[k0+kk][n0+n] ---
    {
      const int n  = tid & 63;
      const int kb = (tid >> 6) * 8;
      #pragma unroll
      for (int j = 0; j < 8; ++j)
        Bsh[n][kb + j] = (_Float16)W[(size_t)(k0 + kb + j) * DD + n0 + n];
    }
    __syncthreads();

    H16 a0, a1;
    {
      const _Float16* ar0 = &Ash[w * 32 + r][0];
      const _Float16* ar1 = &Ash[w * 32 + 16 + r][0];
      a0.h[0] = *(const v8h*)(ar0 + hi * 8);
      a0.h[1] = *(const v8h*)(ar0 + 16 + hi * 8);
      a1.h[0] = *(const v8h*)(ar1 + hi * 8);
      a1.h[1] = *(const v8h*)(ar1 + 16 + hi * 8);
    }
    #pragma unroll
    for (int t = 0; t < 4; ++t) {
      H16 bb;
      const _Float16* br = &Bsh[t * 16 + r][0];
      bb.h[0] = *(const v8h*)(br + hi * 8);
      bb.h[1] = *(const v8h*)(br + 16 + hi * 8);
      acc[0][t] = WMMA_F16(a0.v, bb.v, acc[0][t]);
      acc[1][t] = WMMA_F16(a1.v, bb.v, acc[1][t]);
    }
    __syncthreads();
  }

  // --- epilogue: bias, +P (K/V) or scale (Q), write f16 [B,H,M,HD] ---
  #pragma unroll
  for (int aa = 0; aa < 2; ++aa) {
    #pragma unroll
    for (int v = 0; v < 8; ++v) {
      const int row = r0 + w * 32 + aa * 16 + v + 8 * hi;
      const int b   = row >> 10;
      const int mrw = row & 1023;
      #pragma unroll
      for (int t = 0; t < 4; ++t) {
        const int n  = n0 + t * 16 + r;
        const int h  = n >> 6;
        const int hd = n & 63;
        float val = acc[aa][t][v] + bias[n];
        if (MODE != 0) val += P[mrw * HDD + hd];
        if (MODE == 0) val *= 0.125f;
        Y[(((size_t)b * HH + h) * MM + mrw) * HDD + hd] = (_Float16)val;
      }
    }
  }
}

// ---------------------------------------------------------------------------
// Kernel 3: flash attention per (b,h). 1 wave = 16 queries x 64 head-dim.
// Key blocks of 32, double-buffered in LDS; K staged with async global->LDS
// (ASYNCcnt), V prefetched to VGPRs and transposed into LDS behind the WMMAs.
// ---------------------------------------------------------------------------
__global__ __launch_bounds__(256) void attn_kernel(
    const _Float16* __restrict__ Qf, const _Float16* __restrict__ Kf,
    const _Float16* __restrict__ Vf, const unsigned char* __restrict__ mask,
    _Float16* __restrict__ Ctx)
{
  __shared__ _Float16 Ksh[2][32][64];   // keys x hd, double buffered
  __shared__ _Float16 Vt[2][64][32];    // hd x keys (transposed), double buffered
  __shared__ _Float16 Pl[8][16][32];    // per-wave prob-tile scratch

  const int tid  = threadIdx.x;
  const int w    = tid >> 5;
  const int lane = tid & 31;
  const int r    = lane & 15;
  const int hi   = lane >> 4;
  const int bh   = blockIdx.x >> 3;                // b*8 + h
  const int b    = bh >> 3;
  const int h    = bh & 7;
  const int q0   = ((blockIdx.x & 7) << 7) + (w << 4);

  const _Float16* Qbh = Qf + (size_t)bh * MM * HDD;
  const _Float16* Kbh = Kf + (size_t)bh * MM * HDD;
  const _Float16* Vbh = Vf + (size_t)bh * MM * HDD;

  // staging coordinates (same for every block of 32 keys)
  const int skey = tid >> 3;            // 0..31
  const int sdc  = (tid & 7) * 8;       // 0..56

  // Q A-operands (two K=32 chunks), resident across the whole key loop.
  H16 qa0, qa1;
  {
    const _Float16* qr = Qbh + (size_t)(q0 + r) * HDD;
    qa0.h[0] = *(const v8h*)(qr + hi * 8);
    qa0.h[1] = *(const v8h*)(qr + 16 + hi * 8);
    qa1.h[0] = *(const v8h*)(qr + 32 + hi * 8);
    qa1.h[1] = *(const v8h*)(qr + 48 + hi * 8);
  }

  v8f O0 = {}, O1 = {}, O2 = {}, O3 = {};
  float m[8], l[8];
  #pragma unroll
  for (int v = 0; v < 8; ++v) { m[v] = -3.0e38f; l[v] = 0.0f; }

  // ---- prologue: stage block 0 into buffer 0 ----
  {
    async_lds_b128(&Ksh[0][skey][sdc], Kbh + (size_t)skey * HDD + sdc);
    v8h g = *(const v8h*)(Vbh + (size_t)skey * HDD + sdc);
    #pragma unroll
    for (int j = 0; j < 8; ++j) Vt[0][sdc + j][skey] = g[j];
    asm volatile("s_wait_asynccnt 0" ::: "memory");
  }
  __syncthreads();

  const int NB = MM / 32;
  for (int kb = 0; kb < NB; ++kb) {
    const int bufc = kb & 1;
    const int bufn = bufc ^ 1;
    const int k0   = kb * 32;
    const bool pre = (kb + 1 < NB);

    // --- prefetch next block: async K -> LDS, V -> VGPRs ---
    v8h gv = {};
    if (pre) {
      const size_t nb = (size_t)(k0 + 32 + skey) * HDD + sdc;
      async_lds_b128(&Ksh[bufn][skey][sdc], Kbh + nb);
      gv = *(const v8h*)(Vbh + nb);
    }

    // --- S = Q @ K^T : two 16x16 tiles (keys 0-15 and 16-31) ---
    v8f s0 = {}, s1 = {};
    {
      H16 bb;
      const _Float16* kr0 = &Ksh[bufc][r][0];
      const _Float16* kr1 = &Ksh[bufc][16 + r][0];
      bb.h[0] = *(const v8h*)(kr0 + hi * 8);      bb.h[1] = *(const v8h*)(kr0 + 16 + hi * 8);
      s0 = WMMA_F16(qa0.v, bb.v, s0);
      bb.h[0] = *(const v8h*)(kr0 + 32 + hi * 8); bb.h[1] = *(const v8h*)(kr0 + 48 + hi * 8);
      s0 = WMMA_F16(qa1.v, bb.v, s0);
      bb.h[0] = *(const v8h*)(kr1 + hi * 8);      bb.h[1] = *(const v8h*)(kr1 + 16 + hi * 8);
      s1 = WMMA_F16(qa0.v, bb.v, s1);
      bb.h[0] = *(const v8h*)(kr1 + 32 + hi * 8); bb.h[1] = *(const v8h*)(kr1 + 48 + hi * 8);
      s1 = WMMA_F16(qa1.v, bb.v, s1);
    }

    // --- mask + online softmax (row = v + 8*hi, col group = 16 lanes) ---
    const bool mk0 = mask[b * MM + k0 + r] != 0;
    const bool mk1 = mask[b * MM + k0 + 16 + r] != 0;
    float p0[8], p1[8], sc[8];
    #pragma unroll
    for (int v = 0; v < 8; ++v) {
      float a0 = mk0 ? s0[v] : -3.0e38f;
      float a1 = mk1 ? s1[v] : -3.0e38f;
      float t = fmaxf(a0, a1);
      t = fmaxf(t, __shfl_xor(t, 1, 32));
      t = fmaxf(t, __shfl_xor(t, 2, 32));
      t = fmaxf(t, __shfl_xor(t, 4, 32));
      t = fmaxf(t, __shfl_xor(t, 8, 32));
      const float mn = fmaxf(m[v], t);
      sc[v] = __expf(m[v] - mn);
      p0[v] = __expf(a0 - mn);
      p1[v] = __expf(a1 - mn);
      float rs = p0[v] + p1[v];
      rs += __shfl_xor(rs, 1, 32);
      rs += __shfl_xor(rs, 2, 32);
      rs += __shfl_xor(rs, 4, 32);
      rs += __shfl_xor(rs, 8, 32);
      l[v] = l[v] * sc[v] + rs;
      m[v] = mn;
    }
    #pragma unroll
    for (int v = 0; v < 8; ++v) {
      O0[v] *= sc[v]; O1[v] *= sc[v]; O2[v] *= sc[v]; O3[v] *= sc[v];
    }

    // --- C-layout -> A-layout via per-wave LDS scratch ---
    #pragma unroll
    for (int v = 0; v < 8; ++v) {
      Pl[w][v + 8 * hi][r]      = (_Float16)p0[v];
      Pl[w][v + 8 * hi][16 + r] = (_Float16)p1[v];
    }
    asm volatile("s_wait_dscnt 0" ::: "memory");   // same-wave DS RAW
    H16 pa;
    pa.h[0] = *(const v8h*)(&Pl[w][r][hi * 8]);
    pa.h[1] = *(const v8h*)(&Pl[w][r][16 + hi * 8]);

    // --- O += P @ V : 4 hd tiles ---
    {
      H16 bb;
      const _Float16* vr;
      vr = &Vt[bufc][r][0];
      bb.h[0] = *(const v8h*)(vr + hi * 8); bb.h[1] = *(const v8h*)(vr + 16 + hi * 8);
      O0 = WMMA_F16(pa.v, bb.v, O0);
      vr = &Vt[bufc][16 + r][0];
      bb.h[0] = *(const v8h*)(vr + hi * 8); bb.h[1] = *(const v8h*)(vr + 16 + hi * 8);
      O1 = WMMA_F16(pa.v, bb.v, O1);
      vr = &Vt[bufc][32 + r][0];
      bb.h[0] = *(const v8h*)(vr + hi * 8); bb.h[1] = *(const v8h*)(vr + 16 + hi * 8);
      O2 = WMMA_F16(pa.v, bb.v, O2);
      vr = &Vt[bufc][48 + r][0];
      bb.h[0] = *(const v8h*)(vr + hi * 8); bb.h[1] = *(const v8h*)(vr + 16 + hi * 8);
      O3 = WMMA_F16(pa.v, bb.v, O3);
    }

    // --- finish next-block staging behind the WMMAs ---
    if (pre) {
      #pragma unroll
      for (int j = 0; j < 8; ++j) Vt[bufn][sdc + j][skey] = gv[j];
      asm volatile("s_wait_asynccnt 0" ::: "memory");
    }
    __syncthreads();
  }

  // --- normalize and write ctx f16 in [B, M, H*HD] layout for the out-proj ---
  float inv[8];
  #pragma unroll
  for (int v = 0; v < 8; ++v) inv[v] = (l[v] > 0.0f) ? 1.0f / l[v] : 0.0f;
  _Float16* outBase = Ctx + (size_t)b * MM * DD + (size_t)h * HDD;
  #pragma unroll
  for (int v = 0; v < 8; ++v) {
    const int row = q0 + v + 8 * hi;
    _Float16* orow = outBase + (size_t)row * DD;
    orow[0 * 16 + r] = (_Float16)(O0[v] * inv[v]);
    orow[1 * 16 + r] = (_Float16)(O1[v] * inv[v]);
    orow[2 * 16 + r] = (_Float16)(O2[v] * inv[v]);
    orow[3 * 16 + r] = (_Float16)(O3[v] * inv[v]);
  }
}

// ---------------------------------------------------------------------------
// Kernel 4: output projection  Out = ctx[8192,512](f16) @ Wo + bo  -> f32.
// Block tile 256x64; wave tile 32x64; A read straight from global (f16).
// ---------------------------------------------------------------------------
__global__ __launch_bounds__(256) void oproj_kernel(
    const _Float16* __restrict__ A, const float* __restrict__ W,
    const float* __restrict__ bias, float* __restrict__ Out)
{
  __shared__ _Float16 Bsh[64][32];

  const int tid  = threadIdx.x;
  const int w    = tid >> 5;
  const int lane = tid & 31;
  const int r    = lane & 15;
  const int hi   = lane >> 4;
  const int r0   = blockIdx.x * 256;
  const int n0   = blockIdx.y * 64;

  v8f acc[2][4] = {{v8f{}, v8f{}, v8f{}, v8f{}}, {v8f{}, v8f{}, v8f{}, v8f{}}};
  const _Float16* arow0 = A + (size_t)(r0 + w * 32 + r) * DD;
  const _Float16* arow1 = A + (size_t)(r0 + w * 32 + 16 + r) * DD;

  for (int k0 = 0; k0 < DD; k0 += 32) {
    {
      const int n  = tid & 63;
      const int kb = (tid >> 6) * 8;
      #pragma unroll
      for (int j = 0; j < 8; ++j)
        Bsh[n][kb + j] = (_Float16)W[(size_t)(k0 + kb + j) * DD + n0 + n];
    }
    __syncthreads();

    H16 a0, a1;
    a0.h[0] = *(const v8h*)(arow0 + k0 + hi * 8);
    a0.h[1] = *(const v8h*)(arow0 + k0 + 16 + hi * 8);
    a1.h[0] = *(const v8h*)(arow1 + k0 + hi * 8);
    a1.h[1] = *(const v8h*)(arow1 + k0 + 16 + hi * 8);
    #pragma unroll
    for (int t = 0; t < 4; ++t) {
      H16 bb;
      const _Float16* br = &Bsh[t * 16 + r][0];
      bb.h[0] = *(const v8h*)(br + hi * 8);
      bb.h[1] = *(const v8h*)(br + 16 + hi * 8);
      acc[0][t] = WMMA_F16(a0.v, bb.v, acc[0][t]);
      acc[1][t] = WMMA_F16(a1.v, bb.v, acc[1][t]);
    }
    __syncthreads();
  }

  #pragma unroll
  for (int aa = 0; aa < 2; ++aa) {
    #pragma unroll
    for (int v = 0; v < 8; ++v) {
      const int row = r0 + w * 32 + aa * 16 + v + 8 * hi;
      #pragma unroll
      for (int t = 0; t < 4; ++t) {
        const int n = n0 + t * 16 + r;
        Out[(size_t)row * DD + n] = acc[aa][t][v] + bias[n];
      }
    }
  }
}

// ---------------------------------------------------------------------------
// Host launcher
// ---------------------------------------------------------------------------
extern "C" void kernel_launch(void* const* d_in, const int* in_sizes, int n_in,
                              void* d_out, int out_size, void* d_ws, size_t ws_size,
                              hipStream_t stream) {
  (void)in_sizes; (void)n_in; (void)out_size; (void)ws_size;

  const float* k  = (const float*)d_in[0];
  const float* v  = (const float*)d_in[1];
  const float* q  = (const float*)d_in[2];
  const unsigned char* mask = (const unsigned char*)d_in[3];
  const float* pe = (const float*)d_in[4];
  const float* Wk = (const float*)d_in[5];
  const float* bk = (const float*)d_in[6];
  const float* Wv = (const float*)d_in[7];
  const float* bv = (const float*)d_in[8];
  const float* Wq = (const float*)d_in[9];
  const float* bq = (const float*)d_in[10];
  const float* Wo = (const float*)d_in[11];
  const float* bo = (const float*)d_in[12];
  float* out = (float*)d_out;

  char* ws = (char*)d_ws;
  float* P = (float*)ws;              ws += (size_t)MM * HDD * sizeof(float);
  _Float16* Qf = (_Float16*)ws;       ws += (size_t)BB * HH * MM * HDD * sizeof(_Float16);
  _Float16* Kf = (_Float16*)ws;       ws += (size_t)BB * HH * MM * HDD * sizeof(_Float16);
  _Float16* Vf = (_Float16*)ws;       ws += (size_t)BB * HH * MM * HDD * sizeof(_Float16);
  _Float16* Cf = (_Float16*)ws;       ws += (size_t)BB * MM * DD * sizeof(_Float16);

  pe_reduce_kernel<<<(MM * HDD) / 1024, 256, 0, stream>>>(pe, P);

  dim3 g((BB * MM) / 256, DD / 64);
  proj_kernel<1><<<g, 256, 0, stream>>>(k, Wk, bk, P, Kf);
  proj_kernel<2><<<g, 256, 0, stream>>>(v, Wv, bv, P, Vf);
  proj_kernel<0><<<g, 256, 0, stream>>>(q, Wq, bq, P, Qf);

  attn_kernel<<<BB * HH * (MM / 128), 256, 0, stream>>>(Qf, Kf, Vf, mask, Cf);

  oproj_kernel<<<g, 256, 0, stream>>>(Cf, Wo, bo, out);
}
